// EncoderLayer_75883482186054
// MI455X (gfx1250) — matrix operand order repaired and checked by reading, not compiled, not run on hardware
//
#include <hip/hip_runtime.h>

// ---------------- constants ----------------
#define B_    8
#define C_    64
#define N_    128
#define T_    288
#define PL    12
#define TS    300           // T_ + PL
#define TP    301           // time-padded length (tpad = t+1, tpad 0 == zero pad)
#define CH13  832           // 13*64 concat channels
#define KCONV 1664          // 832 * 2 taps
#define NPOS  (B_*N_*TS)    // 307200 conv output positions
#define OUT1  (B_*C_*N_*T_) // 18874368 (x_z elements)
#define EPS_  0.01f

typedef _Float16 f16;
typedef __attribute__((ext_vector_type(16))) _Float16 v16h;
typedef __attribute__((ext_vector_type(8)))  _Float16 v8h;
typedef __attribute__((ext_vector_type(8)))  float    v8f;
typedef __attribute__((ext_vector_type(4)))  unsigned int v4u;
typedef __attribute__((ext_vector_type(8)))  int v8i;
typedef __attribute__((ext_vector_type(4)))  int v4i;

#if __has_builtin(__builtin_amdgcn_tensor_load_to_lds)
#define USE_TDM 1
#else
#define USE_TDM 0
#endif

__device__ __forceinline__ v8f wmma32(v16h a, v16h b, v8f c) {
    return __builtin_amdgcn_wmma_f32_16x16x32_f16(false, a, false, b, (short)0, c, false, false);
}

// 16 contiguous halves (B-fragment: K = base .. base+15), 16B-aligned
__device__ __forceinline__ v16h ld16c(const f16* p) {
    v8h lo = *(const v8h*)p;
    v8h hi = *(const v8h*)(p + 8);
    return __builtin_shufflevector(lo, hi, 0,1,2,3,4,5,6,7,8,9,10,11,12,13,14,15);
}
// A-fragment, contiguous K order matching B: K = 16g .. 16g+15
__device__ __forceinline__ v16h ldA(const f16* row, int kb, int g) {
    v8h lo = *(const v8h*)(row + kb + 16 * g);
    v8h hi = *(const v8h*)(row + kb + 16 * g + 8);
    return __builtin_shufflevector(lo, hi, 0,1,2,3,4,5,6,7,8,9,10,11,12,13,14,15);
}

__device__ __forceinline__ float wred(float v) {
    #pragma unroll
    for (int m = 16; m >= 1; m >>= 1) v += __shfl_xor(v, m, 32);
    return v;
}

// xcat element index: channel-innermost (b,n,tpad,ch)
__device__ __forceinline__ size_t xidx(int bn, int tp, int ch) {
    return ((size_t)bn * TP + tp) * CH13 + ch;
}

// ---------------- kernel 1: weight convert/repack to fp16 ----------------
__global__ void k_wcvt(const float* w0, const float* w1, const float* w2, const float* w3,
                       const float* c1w, const float* c2w, const float* rw, const float* sw,
                       f16* wre, f16* wconv, f16* wres, f16* wskip) {
    int tid = blockIdx.x * blockDim.x + threadIdx.x;
    const int RE = 4 * 768 * 768;
    const int CV = 128 * KCONV;
    if (tid < RE) {
        int r = tid / (768 * 768), j = tid % (768 * 768);
        const float* src = (r == 0) ? w0 : (r == 1) ? w1 : (r == 2) ? w2 : w3;
        wre[tid] = (f16)src[j];                  // already o*768 + ci*12 + l
    } else if (tid < RE + CV) {
        int j = tid - RE;
        int o = j / KCONV, k = j % KCONV;
        int tap = (k >= 832) ? 1 : 0;
        int cc = k - 832 * tap;
        const float* src = (o < 64) ? c1w : c2w;
        wconv[j] = (f16)src[(size_t)(o & 63) * KCONV + cc * 2 + tap];
    } else if (tid < RE + CV + 4096) {
        wres[tid - RE - CV] = (f16)rw[tid - RE - CV];
    } else if (tid < RE + CV + 8192) {
        wskip[tid - RE - CV - 4096] = (f16)sw[tid - RE - CV - 4096];
    }
}

// ---------------- kernel 2: softmax adjacency ----------------
__device__ __forceinline__ float dotrow(const float* emb, int n, int m) {
    float d = 0.f;
    for (int k = 0; k < 64; ++k) d += emb[n * 64 + k] * emb[m * 64 + k];
    if (m == n) d -= 10.f;
    return d;
}
__global__ void k_adj(const float* emb, float* adj) {
    int n = threadIdx.x;
    if (n >= 128) return;
    float mx = -1e30f;
    for (int m = 0; m < 128; ++m) mx = fmaxf(mx, dotrow(emb, n, m));
    float s = 0.f;
    for (int m = 0; m < 128; ++m) s += expf(dotrow(emb, n, m) - mx);
    for (int m = 0; m < 128; ++m) adj[n * 128 + m] = expf(dotrow(emb, n, m) - mx) / s;
}

// ---------------- kernel 3: per-row norms (global, seasonal, short) ----------------
__global__ void k_stage1(const float* x, f16* xcat, float* X3,
                         f16* xl1, f16* xl2, f16* xl3) {
    __shared__ float L[8][288];
    __shared__ float pm[8][24], pr[8][24], psd[8][24];
    int lane = threadIdx.x & 31, w = threadIdx.x >> 5;
    int r = blockIdx.x * 8 + w;              // (b*64 + c)*128 + n
    int n = r & 127, c = (r >> 7) & 63, b = r >> 13;
    int bn = b * 128 + n;
    const float* xr = x + (size_t)r * T_;
    float* Lr = L[w];
    size_t base = (size_t)bn * TP * CH13;    // + tp*832 + ch

    float s = 0.f, s2 = 0.f;
    for (int t = lane; t < T_; t += 32) { float v = xr[t]; Lr[t] = v; s += v; s2 += v * v; }
    s = wred(s); s2 = wred(s2);
    float mean0 = s / 288.f;
    float var0  = s2 / 288.f - mean0 * mean0 + 1e-5f;
    float std0  = sqrtf(var0), rd0 = rsqrtf(var0 + EPS_);

    __syncthreads();
    for (int t = lane; t < TS; t += 32) {
        float v = (t < T_) ? Lr[t] : Lr[287];
        size_t o = base + (size_t)(t + 1) * CH13 + c;
        xcat[o + 0 * 64] = (f16)v;
        xcat[o + 2 * 64] = (f16)mean0;
        xcat[o + 3 * 64] = (f16)std0;
    }
    if (lane == 0) {
        #pragma unroll
        for (int ss = 0; ss < 10; ++ss) xcat[base + ss * 64 + c] = (f16)0.f;  // tpad 0 zero (streams 0..9)
    }
    __syncthreads();
    for (int t = lane; t < T_; t += 32) Lr[t] = (Lr[t] - mean0) * rd0;  // x1
    __syncthreads();
    for (int t = lane; t < T_; t += 32)
        xcat[base + (size_t)(t + 1) * CH13 + 1 * 64 + c] = (f16)Lr[t];
    if (lane < 12) xl1[(size_t)bn * 768 + c * 12 + lane] = (f16)Lr[276 + lane];

    // seasonal (p = 24, 12 chunks)
    if (lane < 24) {
        float ms = 0.f, ssum = 0.f;
        for (int k = 0; k < 12; ++k) { float v = Lr[k * 24 + lane]; ms += v; ssum += v * v; }
        float mn = ms / 12.f, vr = ssum / 12.f - mn * mn + 1e-5f;
        pm[w][lane] = mn; pr[w][lane] = rsqrtf(vr + EPS_); psd[w][lane] = sqrtf(vr);
    }
    __syncthreads();
    for (int t = lane; t < TS; t += 32) {
        int ph = t % 24;
        size_t o = base + (size_t)(t + 1) * CH13 + c;
        xcat[o + 5 * 64] = (f16)pm[w][ph];
        xcat[o + 6 * 64] = (f16)psd[w][ph];
    }
    for (int t = lane; t < T_; t += 32) { int ph = t % 24; Lr[t] = (Lr[t] - pm[w][ph]) * pr[w][ph]; } // x2
    __syncthreads();
    for (int t = lane; t < T_; t += 32)
        xcat[base + (size_t)(t + 1) * CH13 + 4 * 64 + c] = (f16)Lr[t];
    if (lane < 12) xl2[(size_t)bn * 768 + c * 12 + lane] = (f16)Lr[276 + lane];
    __syncthreads();

    // short-window norm (p = 12, left-edge replicated)
    float mv[9], sv[9], xv[9];
    #pragma unroll
    for (int i = 0; i < 9; ++i) {
        int t = lane + 32 * i;
        int tt = (t < 11) ? 11 : t;
        float ms = 0.f, ssum = 0.f;
        for (int j = 0; j < 12; ++j) { float v = Lr[tt - 11 + j]; ms += v; ssum += v * v; }
        float mn = ms / 12.f, vr = ssum / 12.f - mn * mn + 1e-5f;
        mv[i] = mn; sv[i] = sqrtf(vr);
        xv[i] = (Lr[t] - mn) * rsqrtf(vr + EPS_);
    }
    __syncthreads();
    float* X3r = X3 + (size_t)r * T_;
    #pragma unroll
    for (int i = 0; i < 9; ++i) {
        int t = lane + 32 * i;
        Lr[t] = xv[i]; X3r[t] = xv[i];
        xcat[base + (size_t)(t + 1) * CH13 + 7 * 64 + c] = (f16)xv[i];
    }
    float m287 = __shfl(mv[8], 31, 32);
    float sd287 = __shfl(sv[8], 31, 32);
    {
        int i = 0;
        for (int t = lane; t < TS; t += 32, ++i) {
            float mval, sval;
            if (t < T_) { mval = mv[i]; sval = sv[i]; } else { mval = m287; sval = sd287; }
            size_t o = base + (size_t)(t + 1) * CH13 + c;
            xcat[o + 8 * 64] = (f16)mval;
            xcat[o + 9 * 64] = (f16)sval;
        }
    }
    __syncthreads();
    if (lane < 12) xl3[(size_t)bn * 768 + c * 12 + lane] = (f16)Lr[276 + lane];
}

// ---------------- kernel 4: spatial norm (adjacency mix over nodes) ----------------
__global__ void k_spatial(const float* X3, const float* adj, f16* xcat, f16* xl4) {
    int bc = blockIdx.x;                 // b*64 + c
    int t0 = blockIdx.y * 48;
    int b = bc >> 6, c = bc & 63;
    __shared__ float Sc[128 * 48];
    const float* Xb = X3 + (size_t)bc * N_ * T_;
    for (int i = threadIdx.x; i < 128 * 48; i += 256) {
        int nn = i / 48, tl = i % 48;
        Sc[i] = Xb[(size_t)nn * T_ + t0 + tl];
    }
    __syncthreads();
    for (int u = threadIdx.x; u < 128 * 48; u += 256) {
        int n = u / 48, tl = u % 48, t = t0 + tl;
        float am = 0.f, av = 0.f;
        const float* ar = adj + n * 128;
        for (int m = 0; m < 128; ++m) {
            float a = ar[m], v = Sc[m * 48 + tl];
            am += a * v; av += a * v * v;
        }
        float var = av - am * am + 1e-5f;
        float sd = sqrtf(var), rd = rsqrtf(var + EPS_);
        float x4 = (Sc[n * 48 + tl] - am) * rd;
        int bn = b * 128 + n;
        size_t bb = (size_t)bn * TP * CH13 + c;
        size_t o = bb + (size_t)(t + 1) * CH13;
        xcat[o + 10 * 64] = (f16)x4;
        xcat[o + 11 * 64] = (f16)am;
        xcat[o + 12 * 64] = (f16)sd;
        if (t == 0) {
            xcat[bb + 10 * 64] = (f16)0.f;
            xcat[bb + 11 * 64] = (f16)0.f;
            xcat[bb + 12 * 64] = (f16)0.f;
        }
        if (t == 287) {
            for (int q = 0; q < 12; ++q) {
                size_t op = bb + (size_t)(289 + q) * CH13;
                xcat[op + 11 * 64] = (f16)am;
                xcat[op + 12 * 64] = (f16)sd;
            }
        }
        if (t >= 276) xl4[(size_t)bn * 768 + c * 12 + (t - 276)] = (f16)x4;
    }
}

// ---------------- kernel 5: residual-extrap GEMM (WMMA), scatter to xcat ----------------
__global__ void k_proj(const f16* __restrict__ A, const f16* __restrict__ Bm,
                       const float* __restrict__ bias, f16* xcat, int sidx) {
    int lane = threadIdx.x & 31, ws = threadIdx.x >> 5;
    int w = blockIdx.x * 8 + ws;         // 0..3071
    int mt = w >> 6, nt = w & 63;        // 48 M-tiles x 64 N-tiles
    int cl = lane & 15, g = lane >> 4;
    v8f acc; for (int i = 0; i < 8; ++i) acc[i] = 0.f;
    const f16* arow = A + (size_t)(mt * 16 + cl) * 768;
    const f16* brow = Bm + (size_t)(nt * 16 + cl) * 768;   // pos-major, K innermost
    for (int kk = 0; kk < 24; ++kk) {
        int kb = kk * 32;
        v16h fa = ldA(arow, kb, g);
        v16h fb = ld16c(brow + kb + 16 * g);
        acc = wmma32(fa, fb, acc);
    }
    #pragma unroll
    for (int v = 0; v < 8; ++v) {
        int o = mt * 16 + v + 8 * g;             // 0..767 (= p*64 + c)
        float val = acc[v] + bias[o];
        int c = o & 63, pq = o >> 6;
        int pos = nt * 16 + cl;                  // = b*128 + n
        xcat[xidx(pos, 289 + pq, sidx * 64 + c)] = (f16)val;
    }
}

// ---------------- kernel 6: main gated conv GEMM (WMMA + TDM double-buffered weights) ----------------
__global__ void k_conv(const f16* __restrict__ W, const f16* __restrict__ xcat,
                       const float* __restrict__ b1, const float* __restrict__ b2,
                       f16* __restrict__ Z) {
#if USE_TDM
    __shared__ __align__(16) f16 As[2 * 128 * 128];   // double-buffered 128x128 K-slab (64KB)
#else
    __shared__ __align__(16) f16 As[128 * 128];       // single 32KB slab
#endif
    int tid = threadIdx.x;
    int lane = tid & 31, ws = tid >> 5;
    int wv = blockIdx.x * 8 + ws;        // 0..9599; two 16-wide N-tiles per wave
    int cl = lane & 15, g = lane >> 4;
    int pos0 = wv * 32 + cl;
    int pos1 = pos0 + 16;
    int bn0 = pos0 / TS, t0 = pos0 % TS;
    int bn1 = pos1 / TS, t1 = pos1 % TS;
    v8f acc0[8], acc1[8];
    #pragma unroll
    for (int m = 0; m < 8; ++m)
        #pragma unroll
        for (int i = 0; i < 8; ++i) { acc0[m][i] = 0.f; acc1[m][i] = 0.f; }

#if USE_TDM
    // Tensor Data Mover: DMA one 128(row) x 128(K) fp16 weight slab into LDS buffer (ks&1).
    // As is the only LDS object in this kernel -> its LDS offset is 0.
    auto issue_slab = [&](int ks) {
        unsigned lds_off = (unsigned)((ks & 1) * (128 * 128 * 2));
        unsigned long long ga = (unsigned long long)(const void*)(W + (size_t)ks * 128);
        v4u g0;
        g0[0] = 1u;                                   // count=1, user descriptor
        g0[1] = lds_off;                              // lds_addr (bytes)
        g0[2] = (unsigned)ga;                         // global_addr[31:0]
        g0[3] = (unsigned)((ga >> 32) & 0x1FFFFFFull) | (2u << 30);  // addr[56:32] | type=2
        v8i g1;
        g1[0] = 1 << 16;                              // data_size = 1 (2 bytes)
        g1[1] = (int)(1664u << 16);                   // tensor_dim0 = 1664 (low16 at bits 63:48)
        g1[2] = (int)(128u << 16);                    // tensor_dim1 = 128 (low16 at bits 95:80)
        g1[3] = (int)(128u << 16);                    // tile_dim0 = 128 (bits 127:112)
        g1[4] = 128;                                  // tile_dim1 = 128, tile_dim2 = 0
        g1[5] = 1664;                                 // tensor_dim0_stride[31:0]
        g1[6] = 0;                                    // stride hi / dim1_stride lo
        g1[7] = 0;
        v4i gz; gz[0] = 0; gz[1] = 0; gz[2] = 0; gz[3] = 0;
#if __clang_major__ >= 23
        v8i gz8; for (int q = 0; q < 8; ++q) gz8[q] = 0;
        __builtin_amdgcn_tensor_load_to_lds(g0, g1, gz, gz, gz8, 0);
#else
        __builtin_amdgcn_tensor_load_to_lds(g0, g1, gz, gz, 0);
#endif
    };
    if (ws == 0) issue_slab(0);
#endif

    for (int ks = 0; ks < 13; ++ks) {                 // 13 slabs x 128 K = 1664
#if USE_TDM
        if (ws == 0) __builtin_amdgcn_s_wait_tensorcnt(0);  // slab ks resident
        __syncthreads();
        if (ws == 0 && ks < 12) issue_slab(ks + 1);   // overlap DMA of next slab with compute
        const f16* Asl = As + (ks & 1) * (128 * 128);
#else
        __syncthreads();
        {   // manual staging fallback: 64 halves per thread, vectorized
            int o = tid >> 1, seg = (tid & 1) * 64;
            #pragma unroll
            for (int q = 0; q < 8; ++q)
                *(v8h*)(As + o * 128 + seg + q * 8) =
                    *(const v8h*)(W + (size_t)o * KCONV + ks * 128 + seg + q * 8);
        }
        __syncthreads();
        const f16* Asl = As;
#endif
        #pragma unroll
        for (int kc = 0; kc < 4; ++kc) {
            int K0 = ks * 128 + kc * 32 + 16 * g;
            int tap = (K0 >= 832) ? 1 : 0;
            int cc = K0 - 832 * tap;
            v16h fb0 = ld16c(xcat + xidx(bn0, t0 + tap, cc));   // tap0 -> time t-1, tap1 -> time t
            v16h fb1 = ld16c(xcat + xidx(bn1, t1 + tap, cc));
            #pragma unroll
            for (int mt = 0; mt < 8; ++mt) {
                v16h fa = ldA(Asl + (mt * 16 + cl) * 128 + kc * 32, 0, g);
                acc0[mt] = wmma32(fa, fb0, acc0[mt]);
                acc1[mt] = wmma32(fa, fb1, acc1[mt]);
            }
        }
    }
    // z = tanh(conv1) * sigmoid(conv2); packed b128 stores, channel-innermost Z
    f16* zp0 = Z + ((size_t)bn0 * TS + t0) * 64;
    f16* zp1 = Z + ((size_t)bn1 * TS + t1) * 64;
    #pragma unroll
    for (int mt = 0; mt < 4; ++mt) {
        v8h zv0, zv1;
        #pragma unroll
        for (int v = 0; v < 8; ++v) {
            int o = mt * 16 + v + 8 * g;  // 0..63
            float a1 = acc0[mt][v] + b1[o];
            float a2 = acc0[mt + 4][v] + b2[o];
            zv0[v] = (f16)(tanhf(a1) * (1.f / (1.f + expf(-a2))));
            float c1 = acc1[mt][v] + b1[o];
            float c2 = acc1[mt + 4][v] + b2[o];
            zv1[v] = (f16)(tanhf(c1) * (1.f / (1.f + expf(-c2))));
        }
        *(v8h*)(zp0 + mt * 16 + 8 * g) = zv0;
        *(v8h*)(zp1 + mt * 16 + 8 * g) = zv1;
    }
}

// ---------------- kernel 7: residual 1x1 conv over t<288 (WMMA) ----------------
__global__ void k_res(const f16* __restrict__ A, const f16* __restrict__ Z,
                      const float* __restrict__ bias, float* __restrict__ out) {
    int tid = threadIdx.x;
    int lane = tid & 31, ws = tid >> 5;
    int nt = blockIdx.x * 8 + ws;        // 0..18431
    int cl = lane & 15, g = lane >> 4;
    int pos = nt * 16 + cl;              // over bn*288 + t
    int bn = pos / T_, t = pos % T_;
    const f16* zrow = Z + ((size_t)bn * TS + t) * 64;
    v8f acc[4];
    #pragma unroll
    for (int m = 0; m < 4; ++m)
        #pragma unroll
        for (int i = 0; i < 8; ++i) acc[m][i] = 0.f;
    #pragma unroll
    for (int kk = 0; kk < 2; ++kk) {
        v16h fb = ld16c(zrow + kk * 32 + 16 * g);
        #pragma unroll
        for (int mt = 0; mt < 4; ++mt) {
            v16h fa = ldA(A + (mt * 16 + cl) * 64, kk * 32, g);
            acc[mt] = wmma32(fa, fb, acc[mt]);
        }
    }
    int b = bn >> 7, n = bn & 127;
    #pragma unroll
    for (int mt = 0; mt < 4; ++mt)
        #pragma unroll
        for (int v = 0; v < 8; ++v) {
            int c = mt * 16 + v + 8 * g;
            out[((size_t)(b * 64 + c) * 128 + n) * T_ + t] = acc[mt][v] + bias[c];
        }
}

// ---------------- kernel 8: skip 1x1 conv over last 12 steps (WMMA) ----------------
__global__ void k_skip(const f16* __restrict__ A, const f16* __restrict__ Z,
                       const float* __restrict__ bias, float* __restrict__ out) {
    int tid = threadIdx.x;
    int lane = tid & 31, ws = tid >> 5;
    int bn = blockIdx.x * 8 + ws;        // 0..1023, one tile per (b,n)
    int cl = lane & 15, g = lane >> 4;
    int pc = (cl < 12) ? cl : 11;        // clamp dummy columns
    const f16* zrow = Z + ((size_t)bn * TS + 288 + pc) * 64;
    v8f acc[4];
    #pragma unroll
    for (int m = 0; m < 4; ++m)
        #pragma unroll
        for (int i = 0; i < 8; ++i) acc[m][i] = 0.f;
    #pragma unroll
    for (int kk = 0; kk < 2; ++kk) {
        v16h fb = ld16c(zrow + kk * 32 + 16 * g);
        #pragma unroll
        for (int mt = 0; mt < 4; ++mt) {
            v16h fa = ldA(A + (mt * 16 + cl) * 64, kk * 32, g);
            acc[mt] = wmma32(fa, fb, acc[mt]);
        }
    }
    int b = bn >> 7, n = bn & 127;
    if (cl < 12) {
        #pragma unroll
        for (int mt = 0; mt < 4; ++mt)
            #pragma unroll
            for (int v = 0; v < 8; ++v) {
                int c = mt * 16 + v + 8 * g;
                out[(size_t)OUT1 + ((size_t)(b * 64 + c) * 128 + n) * 12 + cl] = acc[mt][v] + bias[c];
            }
    }
}

// ---------------- launch ----------------
extern "C" void kernel_launch(void* const* d_in, const int* in_sizes, int n_in,
                              void* d_out, int out_size, void* d_ws, size_t ws_size,
                              hipStream_t stream) {
    const float* x      = (const float*)d_in[0];
    const float* emb    = (const float*)d_in[1];
    const float* rew[4] = {(const float*)d_in[2], (const float*)d_in[4],
                           (const float*)d_in[6], (const float*)d_in[8]};
    const float* reb[4] = {(const float*)d_in[3], (const float*)d_in[5],
                           (const float*)d_in[7], (const float*)d_in[9]};
    const float* c1w = (const float*)d_in[10];
    const float* c1b = (const float*)d_in[11];
    const float* c2w = (const float*)d_in[12];
    const float* c2b = (const float*)d_in[13];
    const float* skw = (const float*)d_in[14];
    const float* skb = (const float*)d_in[15];
    const float* rsw = (const float*)d_in[16];
    const float* rsb = (const float*)d_in[17];
    float* out = (float*)d_out;

    // workspace carve-up (256B aligned)
    char* base = (char*)d_ws;
    size_t off = 0;
    auto alloc = [&](size_t bytes) -> void* {
        void* r = base + off;
        off = (off + bytes + 255) & ~(size_t)255;
        return r;
    };
    float* adj   = (float*)alloc(128 * 128 * sizeof(float));
    f16*   wre   = (f16*)  alloc((size_t)4 * 768 * 768 * sizeof(f16));
    f16*   wconv = (f16*)  alloc((size_t)128 * KCONV * sizeof(f16));
    f16*   wres  = (f16*)  alloc(64 * 64 * sizeof(f16));
    f16*   wskip = (f16*)  alloc(64 * 64 * sizeof(f16));
    f16*   xlast = (f16*)  alloc((size_t)4 * 1024 * 768 * sizeof(f16));   // pos-major
    float* X3    = (float*)alloc((size_t)B_ * C_ * N_ * T_ * sizeof(float));
    f16*   Z     = (f16*)  alloc((size_t)B_ * N_ * TS * 64 * sizeof(f16)); // (bn,t,c)
    f16*   xcat  = (f16*)  alloc((size_t)B_ * N_ * TP * CH13 * sizeof(f16)); // (bn,tp,ch)
    (void)ws_size; (void)in_sizes; (void)n_in; (void)out_size;

    // 1. weight convert
    {
        int total = 4 * 768 * 768 + 128 * KCONV + 4096 + 4096;
        k_wcvt<<<(total + 255) / 256, 256, 0, stream>>>(rew[0], rew[1], rew[2], rew[3],
                                                        c1w, c2w, rsw, skw,
                                                        wre, wconv, wres, wskip);
    }
    // 2. adjacency
    k_adj<<<1, 128, 0, stream>>>(emb, adj);
    // 3. per-row norms -> streams 0..9, X3, xlast1..3
    k_stage1<<<(B_ * C_ * N_) / 8, 256, 0, stream>>>(x, xcat, X3,
                                                     xlast + (size_t)0 * 1024 * 768,
                                                     xlast + (size_t)1 * 1024 * 768,
                                                     xlast + (size_t)2 * 1024 * 768);
    // 4. spatial norm -> streams 10..12, xlast4
    k_spatial<<<dim3(B_ * C_, T_ / 48), 256, 0, stream>>>(X3, adj, xcat,
                                                          xlast + (size_t)3 * 1024 * 768);
    // 5. residual-extrap WMMA GEMMs -> pad regions of streams 1,4,7,10
    const int sidx[4] = {1, 4, 7, 10};
    for (int i = 0; i < 4; ++i) {
        k_proj<<<384, 256, 0, stream>>>(wre + (size_t)i * 768 * 768,
                                        xlast + (size_t)i * 1024 * 768,
                                        reb[i], xcat, sidx[i]);
    }
    // 6. main gated conv WMMA GEMM (TDM-staged weights) -> Z
    k_conv<<<(NPOS / 32) / 8, 256, 0, stream>>>(wconv, xcat, c1b, c2b, Z);
    // 7. residual 1x1 -> x_z (t<288)
    k_res<<<((B_ * N_ * T_) / 16) / 8, 256, 0, stream>>>(wres, Z, rsb, out);
    // 8. skip 1x1 -> s (last 12 steps)
    k_skip<<<(B_ * N_) / 8, 256, 0, stream>>>(wskip, Z, skb, out);
}
